// DecoderLayer_27745488732661
// MI455X (gfx1250) — compile-verified, matrix operand order checked by reading, not run on hardware
//
#include <hip/hip_runtime.h>
#include <hip/hip_bf16.h>

#define RES 8192
#define KNB 48
#define H   128

typedef __attribute__((ext_vector_type(16))) __bf16 v16bf;
typedef __attribute__((ext_vector_type(8)))  float  v8f;

__device__ __forceinline__ float gelu_exact(float x) {
    return 0.5f * x * (1.0f + erff(x * 0.70710678118654752f));
}

// A-fragment loader: 16x32 bf16 tile, row-major source with given stride.
// Lane l (h = l>>4): row = rowbase + (l&15); K values = kbase + h*8 + {0..7}
// and kbase + 16 + h*8 + {0..7}  (ISA 7.12.2 16-bit A layout).
__device__ __forceinline__ v16bf load_a_frag(const __bf16* buf, int row, int stride,
                                             int kbase, int h) {
    const __bf16* p = buf + row * stride + kbase + h * 8;
    v16bf a;
#pragma unroll
    for (int i = 0; i < 8; ++i) {
        a[i]     = p[i];
        a[8 + i] = p[16 + i];
    }
    return a;
}

// ---------------------------------------------------------------------------
// Pack fp32 [K,N] row-major weight into bf16 WMMA B-fragment tiles.
// Tile (ks, ct) covers K rows [ks*32, ks*32+32) and N cols [ct*16, ct*16+16).
// Within a tile: lane*16 + j;  n = lane&15, h = lane>>4,
// k_local = (j<8 ? h*8 + j : 16 + h*8 + (j-8))   (mirrors the A layout).
// ---------------------------------------------------------------------------
__global__ void pack_w_kernel(const float* __restrict__ W, __bf16* __restrict__ out,
                              int K, int N, int nct) {
    int idx = blockIdx.x * blockDim.x + threadIdx.x;
    if (idx >= K * N) return;
    int tile  = idx >> 9;          // 512 elements per tile
    int rt    = idx & 511;
    int lane  = rt >> 4;
    int j     = rt & 15;
    int h     = lane >> 4;
    int nl    = lane & 15;
    int kl    = (j < 8) ? (h * 8 + j) : (16 + h * 8 + (j - 8));
    int ks    = tile / nct;
    int ct    = tile % nct;
    int k     = ks * 32 + kl;
    int n     = ct * 16 + nl;
    out[idx]  = (__bf16)W[k * N + n];
}

// ---------------------------------------------------------------------------
// Message kernel: one block (8 waves) per residue.
//   build x[48,512] bf16 in LDS -> GEMM1(512) GELU -> GEMM2(128) GELU
//   -> GEMM3(128) -> sum over 48 edges -> +node_h -> LayerNorm -> h_out
// ---------------------------------------------------------------------------
__global__ void __launch_bounds__(256)
msg_kernel(const float* __restrict__ node_h, const float* __restrict__ edge_h,
           const int* __restrict__ edge_idx, const float* __restrict__ seq_emb,
           const float* __restrict__ ar_mask,
           const __bf16* __restrict__ pW1, const __bf16* __restrict__ pW2,
           const __bf16* __restrict__ pW3,
           const float* __restrict__ mb1, const float* __restrict__ mb2,
           const float* __restrict__ mb3,
           const float* __restrict__ g1, const float* __restrict__ b1,
           float* __restrict__ h_out) {
    __shared__ __align__(16) __bf16 xbuf[KNB * 512];   // 49152 B (aliased as msg2)
    __shared__ __align__(16) __bf16 m1buf[KNB * H];    // 12288 B (aliased as red/ln)
    __shared__ int   jbuf[KNB];
    __shared__ float mbuf[KNB];

    const int r    = blockIdx.x;
    const int tid  = threadIdx.x;
    const int lane = tid & 31;
    const int wave = tid >> 5;          // 0..7 -> output column tile
    const int h    = lane >> 4;
    const int nl   = lane & 15;

    if (tid < KNB) {
        jbuf[tid] = edge_idx[r * KNB + tid];
        mbuf[tid] = ar_mask[r * KNB + tid];
    }
    __syncthreads();

    // ---- build concat input x = [h_i, h_j, edge_h, seq_j*mask] as bf16 ----
    for (int idx = tid; idx < KNB * 512; idx += 256) {
        int k = idx >> 9;
        int c = idx & 511;
        float v;
        if (c < 128)       v = node_h[r * H + c];
        else if (c < 256)  v = node_h[jbuf[k] * H + (c - 128)];
        else if (c < 384)  v = edge_h[(r * KNB + k) * H + (c - 256)];
        else               v = seq_emb[jbuf[k] * H + (c - 384)] * mbuf[k];
        xbuf[idx] = (__bf16)v;
    }
    __syncthreads();

    // ---- GEMM1: [48,512] x [512,128], wave owns 16-col tile `wave` ----
    v8f acc0 = {}, acc1 = {}, acc2 = {};
#pragma unroll 4
    for (int ks = 0; ks < 16; ++ks) {
        v16bf b = *(const v16bf*)(pW1 + ((ks * 8 + wave) * 32 + lane) * 16);
        v16bf a0 = load_a_frag(xbuf,  0 + nl, 512, ks * 32, h);
        v16bf a1 = load_a_frag(xbuf, 16 + nl, 512, ks * 32, h);
        v16bf a2 = load_a_frag(xbuf, 32 + nl, 512, ks * 32, h);
        acc0 = __builtin_amdgcn_wmma_f32_16x16x32_bf16(false, a0, false, b, (short)0, acc0, false, false);
        acc1 = __builtin_amdgcn_wmma_f32_16x16x32_bf16(false, a1, false, b, (short)0, acc1, false, false);
        acc2 = __builtin_amdgcn_wmma_f32_16x16x32_bf16(false, a2, false, b, (short)0, acc2, false, false);
    }
    {
        int col = wave * 16 + nl;
        float bias = mb1[col];
#pragma unroll
        for (int v = 0; v < 8; ++v) {
            int rl = v + h * 8;
            m1buf[(0  + rl) * H + col] = (__bf16)gelu_exact(acc0[v] + bias);
            m1buf[(16 + rl) * H + col] = (__bf16)gelu_exact(acc1[v] + bias);
            m1buf[(32 + rl) * H + col] = (__bf16)gelu_exact(acc2[v] + bias);
        }
    }
    __syncthreads();

    // ---- GEMM2: [48,128] x [128,128], write msg2 into xbuf alias ----
    __bf16* m2buf = xbuf;
    acc0 = (v8f){}; acc1 = (v8f){}; acc2 = (v8f){};
#pragma unroll
    for (int ks = 0; ks < 4; ++ks) {
        v16bf b = *(const v16bf*)(pW2 + ((ks * 8 + wave) * 32 + lane) * 16);
        v16bf a0 = load_a_frag(m1buf,  0 + nl, H, ks * 32, h);
        v16bf a1 = load_a_frag(m1buf, 16 + nl, H, ks * 32, h);
        v16bf a2 = load_a_frag(m1buf, 32 + nl, H, ks * 32, h);
        acc0 = __builtin_amdgcn_wmma_f32_16x16x32_bf16(false, a0, false, b, (short)0, acc0, false, false);
        acc1 = __builtin_amdgcn_wmma_f32_16x16x32_bf16(false, a1, false, b, (short)0, acc1, false, false);
        acc2 = __builtin_amdgcn_wmma_f32_16x16x32_bf16(false, a2, false, b, (short)0, acc2, false, false);
    }
    __syncthreads();   // xbuf reads (GEMM1 phase done), m1buf reads done below barrier
    {
        int col = wave * 16 + nl;
        float bias = mb2[col];
#pragma unroll
        for (int v = 0; v < 8; ++v) {
            int rl = v + h * 8;
            m2buf[(0  + rl) * H + col] = (__bf16)gelu_exact(acc0[v] + bias);
            m2buf[(16 + rl) * H + col] = (__bf16)gelu_exact(acc1[v] + bias);
            m2buf[(32 + rl) * H + col] = (__bf16)gelu_exact(acc2[v] + bias);
        }
    }
    __syncthreads();

    // ---- GEMM3: [48,128] x [128,128], keep raw sums ----
    acc0 = (v8f){}; acc1 = (v8f){}; acc2 = (v8f){};
#pragma unroll
    for (int ks = 0; ks < 4; ++ks) {
        v16bf b = *(const v16bf*)(pW3 + ((ks * 8 + wave) * 32 + lane) * 16);
        v16bf a0 = load_a_frag(m2buf,  0 + nl, H, ks * 32, h);
        v16bf a1 = load_a_frag(m2buf, 16 + nl, H, ks * 32, h);
        v16bf a2 = load_a_frag(m2buf, 32 + nl, H, ks * 32, h);
        acc0 = __builtin_amdgcn_wmma_f32_16x16x32_bf16(false, a0, false, b, (short)0, acc0, false, false);
        acc1 = __builtin_amdgcn_wmma_f32_16x16x32_bf16(false, a1, false, b, (short)0, acc1, false, false);
        acc2 = __builtin_amdgcn_wmma_f32_16x16x32_bf16(false, a2, false, b, (short)0, acc2, false, false);
    }
    __syncthreads();

    // ---- sum msg over the 48 edges: lane covers 24 rows of its column ----
    float* red = (float*)m1buf;           // m1buf free now
    {
        float s = 0.f;
#pragma unroll
        for (int v = 0; v < 8; ++v) s += acc0[v] + acc1[v] + acc2[v];
        red[wave * 32 + lane] = s;
    }
    __syncthreads();

    float* lnv = red + 256;
    if (tid < H) {
        int ct = tid >> 4, nn = tid & 15;
        float agg = red[ct * 32 + nn] + red[ct * 32 + 16 + nn] + 48.0f * mb3[tid];
        lnv[tid] = node_h[r * H + tid] + agg;
    }
    __syncthreads();
    if (tid < H) {
        float mu = 0.f;
#pragma unroll 8
        for (int i = 0; i < H; ++i) mu += lnv[i];
        mu *= (1.0f / H);
        float var = 0.f;
#pragma unroll 8
        for (int i = 0; i < H; ++i) { float d = lnv[i] - mu; var += d * d; }
        var *= (1.0f / H);
        float rs = rsqrtf(var + 1e-5f);
        h_out[r * H + tid] = (lnv[tid] - mu) * rs * g1[tid] + b1[tid];
    }
}

// ---------------------------------------------------------------------------
// FF kernel: 16 residues per block (8 waves).
//   h[16,128] -> GEMM(fW1)[16,512] GELU -> GEMM(fW2)[16,128] + h -> LN -> out
// ---------------------------------------------------------------------------
__global__ void __launch_bounds__(256)
ff_kernel(const float* __restrict__ h_in,
          const __bf16* __restrict__ pF1, const __bf16* __restrict__ pF2,
          const float* __restrict__ fb1, const float* __restrict__ fb2,
          const float* __restrict__ g2, const float* __restrict__ b2,
          float* __restrict__ out) {
    __shared__ __align__(16) __bf16 hb[16 * H];       // 4 KB
    __shared__ __align__(16) __bf16 tb[16 * 512];     // 16 KB
    __shared__ __align__(16) float  ob[16 * H];       // 8 KB

    const int r0   = blockIdx.x * 16;
    const int tid  = threadIdx.x;
    const int lane = tid & 31;
    const int wave = tid >> 5;
    const int h    = lane >> 4;
    const int nl   = lane & 15;

    for (int idx = tid; idx < 16 * H; idx += 256)
        hb[idx] = (__bf16)h_in[r0 * H + idx];
    __syncthreads();

    // ---- GEMM1: [16,128] x [128,512]; wave handles col tiles w, w+8, w+16, w+24
#pragma unroll
    for (int cti = 0; cti < 4; ++cti) {
        int ct = wave + cti * 8;
        v8f acc = {};
#pragma unroll
        for (int ks = 0; ks < 4; ++ks) {
            v16bf b = *(const v16bf*)(pF1 + ((ks * 32 + ct) * 32 + lane) * 16);
            v16bf a = load_a_frag(hb, nl, H, ks * 32, h);
            acc = __builtin_amdgcn_wmma_f32_16x16x32_bf16(false, a, false, b, (short)0, acc, false, false);
        }
        int col = ct * 16 + nl;
        float bias = fb1[col];
#pragma unroll
        for (int v = 0; v < 8; ++v)
            tb[(v + h * 8) * 512 + col] = (__bf16)gelu_exact(acc[v] + bias);
    }
    __syncthreads();

    // ---- GEMM2: [16,512] x [512,128]; wave owns col tile `wave` ----
    {
        v8f acc = {};
#pragma unroll 4
        for (int ks = 0; ks < 16; ++ks) {
            v16bf b = *(const v16bf*)(pF2 + ((ks * 8 + wave) * 32 + lane) * 16);
            v16bf a = load_a_frag(tb, nl, 512, ks * 32, h);
            acc = __builtin_amdgcn_wmma_f32_16x16x32_bf16(false, a, false, b, (short)0, acc, false, false);
        }
        int col = wave * 16 + nl;
        float bias = fb2[col];
#pragma unroll
        for (int v = 0; v < 8; ++v) {
            int row = v + h * 8;
            ob[row * H + col] = acc[v] + bias + h_in[(r0 + row) * H + col];
        }
    }
    __syncthreads();

    // ---- LayerNorm per row; wave w handles rows 2w, 2w+1 ----
#pragma unroll
    for (int rr = 0; rr < 2; ++rr) {
        int row = wave * 2 + rr;
        float v0 = ob[row * H + lane * 4 + 0];
        float v1 = ob[row * H + lane * 4 + 1];
        float v2 = ob[row * H + lane * 4 + 2];
        float v3 = ob[row * H + lane * 4 + 3];
        float s = v0 + v1 + v2 + v3;
#pragma unroll
        for (int off = 16; off >= 1; off >>= 1) s += __shfl_xor(s, off, 32);
        float mu = s * (1.0f / H);
        float d0 = v0 - mu, d1 = v1 - mu, d2 = v2 - mu, d3 = v3 - mu;
        float q = d0 * d0 + d1 * d1 + d2 * d2 + d3 * d3;
#pragma unroll
        for (int off = 16; off >= 1; off >>= 1) q += __shfl_xor(q, off, 32);
        float rs = rsqrtf(q * (1.0f / H) + 1e-5f);
        int cb = lane * 4;
        out[(r0 + row) * H + cb + 0] = d0 * rs * g2[cb + 0] + b2[cb + 0];
        out[(r0 + row) * H + cb + 1] = d1 * rs * g2[cb + 1] + b2[cb + 1];
        out[(r0 + row) * H + cb + 2] = d2 * rs * g2[cb + 2] + b2[cb + 2];
        out[(r0 + row) * H + cb + 3] = d3 * rs * g2[cb + 3] + b2[cb + 3];
    }
}

extern "C" void kernel_launch(void* const* d_in, const int* in_sizes, int n_in,
                              void* d_out, int out_size, void* d_ws, size_t ws_size,
                              hipStream_t stream) {
    const float* node_h  = (const float*)d_in[0];
    const float* edge_h  = (const float*)d_in[1];
    const int*   edge_idx= (const int*)  d_in[2];
    const float* seq_emb = (const float*)d_in[3];
    const float* ar_mask = (const float*)d_in[4];
    const float* mW1 = (const float*)d_in[5];
    const float* mb1 = (const float*)d_in[6];
    const float* mW2 = (const float*)d_in[7];
    const float* mb2 = (const float*)d_in[8];
    const float* mW3 = (const float*)d_in[9];
    const float* mb3 = (const float*)d_in[10];
    const float* fW1 = (const float*)d_in[11];
    const float* fb1 = (const float*)d_in[12];
    const float* fW2 = (const float*)d_in[13];
    const float* fb2 = (const float*)d_in[14];
    const float* g1  = (const float*)d_in[15];
    const float* b1  = (const float*)d_in[16];
    const float* g2  = (const float*)d_in[17];
    const float* b2  = (const float*)d_in[18];

    // workspace layout (bf16 packed weights + fp32 intermediate h)
    __bf16* pW1 = (__bf16*)d_ws;            // 512*128
    __bf16* pW2 = pW1 + 512 * 128;          // 128*128
    __bf16* pW3 = pW2 + 128 * 128;          // 128*128
    __bf16* pF1 = pW3 + 128 * 128;          // 128*512
    __bf16* pF2 = pF1 + 128 * 512;          // 512*128
    float*  hws = (float*)(pF2 + 512 * 128);// RES*H fp32

    pack_w_kernel<<<(512 * 128 + 255) / 256, 256, 0, stream>>>(mW1, pW1, 512, 128, 8);
    pack_w_kernel<<<(128 * 128 + 255) / 256, 256, 0, stream>>>(mW2, pW2, 128, 128, 8);
    pack_w_kernel<<<(128 * 128 + 255) / 256, 256, 0, stream>>>(mW3, pW3, 128, 128, 8);
    pack_w_kernel<<<(128 * 512 + 255) / 256, 256, 0, stream>>>(fW1, pF1, 128, 512, 32);
    pack_w_kernel<<<(512 * 128 + 255) / 256, 256, 0, stream>>>(fW2, pF2, 512, 128, 8);

    msg_kernel<<<RES, 256, 0, stream>>>(node_h, edge_h, edge_idx, seq_emb, ar_mask,
                                        pW1, pW2, pW3, mb1, mb2, mb3, g1, b1, hws);
    ff_kernel<<<RES / 16, 256, 0, stream>>>(hws, pF1, pF2, fb1, fb2, g2, b2,
                                            (float*)d_out);
}
